// PtrNetGRU_25658134627077
// MI455X (gfx1250) — compile-verified, matrix operand order checked by reading, not run on hardware
//
#include <hip/hip_runtime.h>
#include <math.h>

#define B_ 256
#define S_ 128
#define F_ 128
#define H_ 512
#define A_ 512
#define C_ 128

typedef __attribute__((ext_vector_type(16))) __bf16 v16bf;
typedef __attribute__((ext_vector_type(8)))  float  v8f;

// ---------------------------------------------------------------------------
// Fragment loader: 16 rows x 32 cols of a row-major f32 matrix -> bf16 frag.
// 16-bit A-matrix 16x32 layout (ISA 7.12.2): lane L holds row (L&15); K-base
// is 0 for lanes 0-15, 8 for lanes 16-31; VGPRs 0-3 = K{base..base+7},
// VGPRs 4-7 = K{base+16..base+23}. B (32x16 from W[N,K] row-major) mirrors
// this with N in place of M. Emits 4x global_load_b128 per fragment.
// ---------------------------------------------------------------------------
__device__ __forceinline__ v16bf load_frag(const float* __restrict__ base,
                                           int ld, int r0, int k0) {
  const int lane = threadIdx.x & 31;
  const int rr   = lane & 15;
  const int kb   = (lane & 16) ? 8 : 0;
  const float* p = base + (size_t)(r0 + rr) * ld + (k0 + kb);
  float4 x0 = *(const float4*)(p);
  float4 x1 = *(const float4*)(p + 4);
  float4 x2 = *(const float4*)(p + 16);
  float4 x3 = *(const float4*)(p + 20);
  v16bf f;
  f[0]=(__bf16)x0.x;  f[1]=(__bf16)x0.y;  f[2]=(__bf16)x0.z;  f[3]=(__bf16)x0.w;
  f[4]=(__bf16)x1.x;  f[5]=(__bf16)x1.y;  f[6]=(__bf16)x1.z;  f[7]=(__bf16)x1.w;
  f[8]=(__bf16)x2.x;  f[9]=(__bf16)x2.y;  f[10]=(__bf16)x2.z; f[11]=(__bf16)x2.w;
  f[12]=(__bf16)x3.x; f[13]=(__bf16)x3.y; f[14]=(__bf16)x3.z; f[15]=(__bf16)x3.w;
  return f;
}

#define WMMA_BF16(a, b, c) \
  __builtin_amdgcn_wmma_f32_16x16x32_bf16(false, (a), false, (b), (short)0, (c), false, false)

__device__ __forceinline__ float sigmoidf_(float x) { return 1.0f / (1.0f + __expf(-x)); }

// ---------------------------------------------------------------------------
// Fused GRU cell step. grid = (B/32, H/64), block = 128 (4 waves).
// Each wave computes a 32x16 tile of h_new (2 M-tiles sharing the 3 gate
// weight fragments -> 6 WMMAs per 5 fragment loads). Accumulates 4 gate
// quantities (r, z, i_n, h_n) in f32 C fragments, then applies the
// sigmoid/tanh gate math directly in the C-fragment register layout.
// ---------------------------------------------------------------------------
__global__ __launch_bounds__(128)
void gru_step_kernel(const float* __restrict__ xt, int ld_x,       // [B,F] rows
                     const float* __restrict__ h_in,               // [B,H]
                     const float* __restrict__ Wih,                // [3H,F]
                     const float* __restrict__ Whh,                // [3H,H]
                     const float* __restrict__ bih,
                     const float* __restrict__ bhh,
                     float* __restrict__ h_out,                    // [B,H]
                     float* __restrict__ h_copy, int ld_c)         // optional
{
  const int wave = threadIdx.x >> 5;
  const int lane = threadIdx.x & 31;
  const int row0 = blockIdx.x * 32;            // two 16-row tiles
  const int row1 = row0 + 16;
  const int col0 = (blockIdx.y * 4 + wave) * 16;

  v8f r0a = {}, z0a = {}, in0 = {}, hn0 = {};  // tile 0 accumulators
  v8f r1a = {}, z1a = {}, in1 = {}, hn1 = {};  // tile 1 accumulators

  // ---- input contribution: K over F ----
  for (int k = 0; k < F_; k += 32) {
    v16bf a0 = load_frag(xt, ld_x, row0, k);
    v16bf a1 = load_frag(xt, ld_x, row1, k);
    v16bf br = load_frag(Wih,               F_, col0, k);
    v16bf bz = load_frag(Wih +     H_ * F_, F_, col0, k);
    v16bf bn = load_frag(Wih + 2 * H_ * F_, F_, col0, k);
    r0a = WMMA_BF16(a0, br, r0a);  r1a = WMMA_BF16(a1, br, r1a);
    z0a = WMMA_BF16(a0, bz, z0a);  z1a = WMMA_BF16(a1, bz, z1a);
    in0 = WMMA_BF16(a0, bn, in0);  in1 = WMMA_BF16(a1, bn, in1);
  }
  // ---- hidden contribution: K over H ----
  for (int k = 0; k < H_; k += 32) {
    v16bf a0 = load_frag(h_in, H_, row0, k);
    v16bf a1 = load_frag(h_in, H_, row1, k);
    v16bf br = load_frag(Whh,               H_, col0, k);
    v16bf bz = load_frag(Whh +     H_ * H_, H_, col0, k);
    v16bf bn = load_frag(Whh + 2 * H_ * H_, H_, col0, k);
    r0a = WMMA_BF16(a0, br, r0a);  r1a = WMMA_BF16(a1, br, r1a);
    z0a = WMMA_BF16(a0, bz, z0a);  z1a = WMMA_BF16(a1, bz, z1a);
    hn0 = WMMA_BF16(a0, bn, hn0);  hn1 = WMMA_BF16(a1, bn, hn1);
  }

  // ---- fused gates + writeback (C layout: M = i + 8*(lane>=16), N = lane&15)
  const int mofs = (lane & 16) ? 8 : 0;
  const int n    = col0 + (lane & 15);
  const float br_ = bih[n]          + bhh[n];
  const float bz_ = bih[H_ + n]     + bhh[H_ + n];
  const float bin = bih[2 * H_ + n];
  const float bhn = bhh[2 * H_ + n];
#pragma unroll
  for (int t = 0; t < 2; ++t) {
    const int   rbase = (t == 0 ? row0 : row1) + mofs;
    const v8f&  ar = t == 0 ? r0a : r1a;
    const v8f&  az = t == 0 ? z0a : z1a;
    const v8f&  ai = t == 0 ? in0 : in1;
    const v8f&  ah = t == 0 ? hn0 : hn1;
#pragma unroll
    for (int i = 0; i < 8; ++i) {
      const int m = rbase + i;
      float r  = sigmoidf_(ar[i] + br_);
      float z  = sigmoidf_(az[i] + bz_);
      float nn = tanhf(ai[i] + bin + r * (ah[i] + bhn));
      float hp = h_in[(size_t)m * H_ + n];
      float hv = (1.0f - z) * nn + z * hp;
      h_out[(size_t)m * H_ + n] = hv;
      if (h_copy) h_copy[(size_t)m * ld_c + n] = hv;
    }
  }
}

// ---------------------------------------------------------------------------
// Generic C[M,N] = A[M,K] @ W[N,K]^T via bf16 WMMA, 32x16 per wave (2 WMMAs
// amortize each weight fragment). grid = (M/32, N/64), block = 128.
// ---------------------------------------------------------------------------
__global__ __launch_bounds__(128)
void gemm_nt_bf16_kernel(const float* __restrict__ Am, const float* __restrict__ W,
                         float* __restrict__ Cm, int M, int N, int K)
{
  const int wave = threadIdx.x >> 5;
  const int lane = threadIdx.x & 31;
  const int row0 = blockIdx.x * 32;
  const int row1 = row0 + 16;
  const int col0 = (blockIdx.y * 4 + wave) * 16;
  v8f acc0 = {}, acc1 = {};
  for (int k = 0; k < K; k += 32) {
    v16bf a0 = load_frag(Am, K, row0, k);
    v16bf a1 = load_frag(Am, K, row1, k);
    v16bf b  = load_frag(W,  K, col0, k);
    acc0 = WMMA_BF16(a0, b, acc0);
    acc1 = WMMA_BF16(a1, b, acc1);
  }
  const int mofs = (lane & 16) ? 8 : 0;
  const int n    = col0 + (lane & 15);
#pragma unroll
  for (int i = 0; i < 8; ++i) {
    Cm[(size_t)(row0 + mofs + i) * N + n] = acc0[i];
    Cm[(size_t)(row1 + mofs + i) * N + n] = acc1[i];
  }
}

// ---------------------------------------------------------------------------
// scores[b,s] = sum_a relu(enc_proj[b,s,a] + hw2[b,a]) * v[a]
// One wave per (b,s); 8 waves per block. enc_proj (67MB) is L2-resident.
// ---------------------------------------------------------------------------
__global__ __launch_bounds__(256)
void attn_score_kernel(const float* __restrict__ enc_proj,
                       const float* __restrict__ hw2,
                       const float* __restrict__ v,
                       float* __restrict__ scores)
{
  const int wid  = blockIdx.x * 8 + (threadIdx.x >> 5);   // (b,s) flat
  const int lane = threadIdx.x & 31;
  const int b    = wid / S_;
  const float* ep = enc_proj + (size_t)wid * A_;
  const float* hb = hw2 + (size_t)b * A_;
  float sum = 0.0f;
  for (int a = lane * 4; a < A_; a += 32 * 4) {
    float4 e  = *(const float4*)(ep + a);
    float4 h4 = *(const float4*)(hb + a);
    float4 vv = *(const float4*)(v + a);
    sum += fmaxf(e.x + h4.x, 0.0f) * vv.x;
    sum += fmaxf(e.y + h4.y, 0.0f) * vv.y;
    sum += fmaxf(e.z + h4.z, 0.0f) * vv.z;
    sum += fmaxf(e.w + h4.w, 0.0f) * vv.w;
  }
#pragma unroll
  for (int off = 16; off; off >>= 1) sum += __shfl_xor(sum, off, 32);
  if (lane == 0) scores[wid] = sum;
}

// ---------------------------------------------------------------------------
// Per-batch: softmax(scores) -> prob, log_softmax(prob) (faithful to ref),
// step loss, argmax pred, gather next decoder input. One block per b, S threads.
// ---------------------------------------------------------------------------
__global__ __launch_bounds__(128)
void softmax_pred_kernel(const float* __restrict__ scores,
                         const int* __restrict__ y, int step,
                         const float* __restrict__ x,
                         float* __restrict__ dec_in,
                         float* __restrict__ preds,     // d_out [B,C]
                         float* __restrict__ loss_acc)
{
  __shared__ float sh[S_];
  __shared__ int   shi[S_];
  __shared__ int   sbest;
  const int b = blockIdx.x, s = threadIdx.x;
  const float sc = scores[b * S_ + s];

  sh[s] = sc; __syncthreads();
  for (int off = 64; off >= 1; off >>= 1) {
    if (s < off) sh[s] = fmaxf(sh[s], sh[s + off]);
    __syncthreads();
  }
  float mx = sh[0]; __syncthreads();
  float e = __expf(sc - mx);
  sh[s] = e; __syncthreads();
  for (int off = 64; off >= 1; off >>= 1) {
    if (s < off) sh[s] += sh[s + off];
    __syncthreads();
  }
  float prob = e / sh[0]; __syncthreads();

  // log_softmax applied to probabilities (faithful to reference)
  sh[s] = prob; __syncthreads();
  for (int off = 64; off >= 1; off >>= 1) {
    if (s < off) sh[s] = fmaxf(sh[s], sh[s + off]);
    __syncthreads();
  }
  float mx2 = sh[0]; __syncthreads();
  float e2 = __expf(prob - mx2);
  sh[s] = e2; __syncthreads();
  for (int off = 64; off >= 1; off >>= 1) {
    if (s < off) sh[s] += sh[s + off];
    __syncthreads();
  }
  float logp = (prob - mx2) - __logf(sh[0]); __syncthreads();

  // argmax (argmax(prob) == argmax(sc)); ties keep first index
  sh[s] = sc; shi[s] = s; __syncthreads();
  for (int off = 64; off >= 1; off >>= 1) {
    if (s < off && sh[s + off] > sh[s]) { sh[s] = sh[s + off]; shi[s] = shi[s + off]; }
    __syncthreads();
  }
  if (s == 0) {
    sbest = shi[0];
    preds[(size_t)b * C_ + step] = (float)shi[0];
  }
  if (s == y[b * C_ + step]) atomicAdd(loss_acc, -logp * (1.0f / B_));
  __syncthreads();
  const int best = sbest;
  for (int f = s; f < F_; f += S_)
    dec_in[(size_t)b * F_ + f] = x[((size_t)b * S_ + best) * F_ + f];
}

__global__ void finalize_loss_kernel(const float* __restrict__ loss_acc,
                                     float* __restrict__ out) {
  out[0] = loss_acc[0] * (1.0f / ((float)B_ * (float)C_));
}

// ---------------------------------------------------------------------------
extern "C" void kernel_launch(void* const* d_in, const int* in_sizes, int n_in,
                              void* d_out, int out_size, void* d_ws, size_t ws_size,
                              hipStream_t stream) {
  const float* x       = (const float*)d_in[0];
  const int*   y       = (const int*)  d_in[1];
  const float* enc_Wih = (const float*)d_in[2];
  const float* enc_Whh = (const float*)d_in[3];
  const float* enc_bih = (const float*)d_in[4];
  const float* enc_bhh = (const float*)d_in[5];
  const float* dec_Wih = (const float*)d_in[6];
  const float* dec_Whh = (const float*)d_in[7];
  const float* dec_bih = (const float*)d_in[8];
  const float* dec_bhh = (const float*)d_in[9];
  const float* w1      = (const float*)d_in[10];
  const float* w2      = (const float*)d_in[11];
  const float* v       = (const float*)d_in[12];
  float* out = (float*)d_out;

  float* ws = (float*)d_ws;
  size_t o = 0;
  float* h_a      = ws + o; o += (size_t)B_ * H_;
  float* h_b      = ws + o; o += (size_t)B_ * H_;
  float* enc_out  = ws + o; o += (size_t)B_ * S_ * H_;
  float* enc_proj = ws + o; o += (size_t)B_ * S_ * A_;
  float* hw2      = ws + o; o += (size_t)B_ * A_;
  float* scores   = ws + o; o += (size_t)B_ * S_;
  float* dec_in   = ws + o; o += (size_t)B_ * F_;
  float* loss_acc = ws + o; o += 1;

  hipMemsetAsync(h_a,      0, (size_t)B_ * H_ * sizeof(float), stream);
  hipMemsetAsync(dec_in,   0, (size_t)B_ * F_ * sizeof(float), stream);
  hipMemsetAsync(loss_acc, 0, sizeof(float), stream);

  dim3 ggrid(B_ / 32, H_ / 64);          // GRU step: 64 WGs, 256 waves
  dim3 pgrid((B_ * S_) / 32, A_ / 64);   // enc_proj GEMM
  dim3 wgrid(B_ / 32, A_ / 64);          // h @ w2^T GEMM

  // ---- encoder recurrence ----
  float* hc = h_a; float* hn = h_b;
  for (int t = 0; t < S_; ++t) {
    gru_step_kernel<<<ggrid, 128, 0, stream>>>(
        x + (size_t)t * F_, S_ * F_, hc,
        enc_Wih, enc_Whh, enc_bih, enc_bhh,
        hn, enc_out + (size_t)t * H_, S_ * H_);
    float* tmp = hc; hc = hn; hn = tmp;
  }

  // ---- hoisted attention projection: enc_proj = enc_out @ w1^T ----
  gemm_nt_bf16_kernel<<<pgrid, 128, 0, stream>>>(enc_out, w1, enc_proj,
                                                 B_ * S_, A_, H_);

  // ---- decoder pointer loop ----
  for (int step = 0; step < C_; ++step) {
    gru_step_kernel<<<ggrid, 128, 0, stream>>>(
        dec_in, F_, hc, dec_Wih, dec_Whh, dec_bih, dec_bhh, hn, nullptr, 0);
    float* tmp = hc; hc = hn; hn = tmp;
    gemm_nt_bf16_kernel<<<wgrid, 128, 0, stream>>>(hc, w2, hw2, B_, A_, H_);
    attn_score_kernel<<<(B_ * S_) / 8, 256, 0, stream>>>(enc_proj, hw2, v, scores);
    softmax_pred_kernel<<<B_, 128, 0, stream>>>(scores, y, step, x, dec_in,
                                                out, loss_acc);
  }
  finalize_loss_kernel<<<1, 1, 0, stream>>>(loss_acc, out + (size_t)B_ * C_);
}
